// Low_Rank_SwinTransformerBlock_81080392614566
// MI455X (gfx1250) — matrix operand order repaired and checked by reading, not compiled, not run on hardware
//
#include <hip/hip_runtime.h>
#include <hip/hip_bf16.h>
#include <math.h>

// ---------------------------------------------------------------------------
// Low-rank shifted-window Swin block for gfx1250 (MI455X).
// GEMMs: v_wmma_f32_16x16x32_bf16, 32x32 per wave, 64x64 per block,
// double-buffered LDS tiles fed by async global->LDS DMA when available.
// ---------------------------------------------------------------------------

typedef __bf16 bf16;
typedef __attribute__((ext_vector_type(16))) __bf16 v16bf;
typedef __attribute__((ext_vector_type(8)))  __bf16 v8bf;
typedef __attribute__((ext_vector_type(8)))  float  v8f;
typedef int v4i __attribute__((vector_size(16)));

union V16 { v16bf v; v8bf h[2]; };

__device__ __forceinline__ v8bf load8g(const bf16* p) {
  return *reinterpret_cast<const v8bf*>(p);
}

// constants
#define BATCH 32
#define HH 56
#define CC 384
#define WS 7
#define NTOK (BATCH * HH * HH)          // 100352
#define MROWS NTOK
#define NHEADS 12
#define HD 32
#define RANK 192
#define HID 1536
#define NWS 49
#define SHIFT_ 3

// ---------------------------------------------------------------------------
// async global -> LDS copy (16B per lane), with safe fallback
// ---------------------------------------------------------------------------
#if __has_builtin(__builtin_amdgcn_global_load_async_to_lds_b128)
#define HAVE_ASYNC_CP 1
#else
#define HAVE_ASYNC_CP 0
#endif

__device__ __forceinline__ void async_cp16(const void* g, void* l) {
#if HAVE_ASYNC_CP
  __builtin_amdgcn_global_load_async_to_lds_b128(
      (__attribute__((address_space(1))) v4i*)g,
      (__attribute__((address_space(3))) v4i*)l, 0, 0);
#else
  *(v8bf*)l = *(const v8bf*)g;
#endif
}

__device__ __forceinline__ void async_wait_all() {
#if HAVE_ASYNC_CP
#if __has_builtin(__builtin_amdgcn_s_wait_asynccnt)
  __builtin_amdgcn_s_wait_asynccnt(0);
#else
  asm volatile("s_wait_asynccnt 0x0" ::: "memory");
#endif
#endif
}

// window-row -> token index (reverse shift + window reverse)
__device__ __forceinline__ int swin_remap(int r) {
  int g = r / NWS, n = r % NWS;
  int b = g >> 6, wh = (g >> 3) & 7, ww = g & 7;
  int i = n / WS, j = n % WS;
  int h = wh * WS + i + SHIFT_; if (h >= HH) h -= HH;
  int w = ww * WS + j + SHIFT_; if (w >= HH) w -= HH;
  return (b * HH + h) * HH + w;
}

// ---------------------------------------------------------------------------
// f32 -> bf16 conversion
// ---------------------------------------------------------------------------
__global__ void cvt_bf16(const float* __restrict__ s, bf16* __restrict__ d, int n) {
  int i = blockIdx.x * 256 + threadIdx.x;
  if (i < n) d[i] = (bf16)s[i];
}

// ---------------------------------------------------------------------------
// fused attention bias: rpb[h] + shift mask[nw], padded 64x64
// ---------------------------------------------------------------------------
__device__ __forceinline__ int region_id(int a) {
  return a < (HH - WS) ? 0 : (a < (HH - SHIFT_) ? 1 : 2);
}

__global__ __launch_bounds__(256)
void attn_bias_kernel(const float* __restrict__ rpb, float* __restrict__ out) {
  int h  = blockIdx.x % NHEADS;
  int nw = blockIdx.x / NHEADS;
  int wh = nw >> 3, ww = nw & 7;
  for (int e = threadIdx.x; e < 64 * 64; e += 256) {
    int n = e >> 6, m = e & 63;
    float v;
    if (m >= NWS) {
      v = -1e30f;                       // padded keys -> softmax weight 0
    } else if (n >= NWS) {
      v = 0.f;                          // padded query rows: unused
    } else {
      int i1 = n / WS, j1 = n % WS, i2 = m / WS, j2 = m % WS;
      int idx = (i1 - i2 + WS - 1) * (2 * WS - 1) + (j1 - j2 + WS - 1);
      v = rpb[idx * NHEADS + h];
      int r1 = region_id(wh * WS + i1) * 3 + region_id(ww * WS + j1);
      int r2 = region_id(wh * WS + i2) * 3 + region_id(ww * WS + j2);
      if (r1 != r2) v -= 100.f;
    }
    out[((size_t)blockIdx.x * 64 + n) * 64 + m] = v;
  }
}

// ---------------------------------------------------------------------------
// LayerNorm (+optional shift/window partition)
// ---------------------------------------------------------------------------
template <bool SWIN>
__global__ __launch_bounds__(128)
void ln_kernel(const float* __restrict__ x, const float* __restrict__ gam,
               const float* __restrict__ bet, bf16* __restrict__ out) {
  int r = blockIdx.x;
  int tok;
  if (SWIN) {
    int g = r / NWS, n = r % NWS;
    int b = g >> 6, wh = (g >> 3) & 7, ww = g & 7;
    int i = n / WS, j = n % WS;
    int h = wh * WS + i + SHIFT_; if (h >= HH) h -= HH;   // roll(-3): src = +3
    int w = ww * WS + j + SHIFT_; if (w >= HH) w -= HH;
    tok = (b * HH + h) * HH + w;
  } else {
    tok = r;
  }
  const float* xr = x + (size_t)tok * CC;
  float s = 0.f, s2 = 0.f;
  for (int c = threadIdx.x; c < CC; c += 128) {
    float v = xr[c]; s += v; s2 += v * v;
  }
  for (int m = 16; m; m >>= 1) {
    s  += __shfl_down(s,  m, 32);
    s2 += __shfl_down(s2, m, 32);
  }
  __shared__ float ls[4], ls2[4];
  int wv = threadIdx.x >> 5, ln = threadIdx.x & 31;
  if (ln == 0) { ls[wv] = s; ls2[wv] = s2; }
  __syncthreads();
  s  = ls[0]  + ls[1]  + ls[2]  + ls[3];
  s2 = ls2[0] + ls2[1] + ls2[2] + ls2[3];
  float mean = s * (1.f / CC);
  float var  = s2 * (1.f / CC) - mean * mean;
  float rstd = rsqrtf(var + 1e-5f);
  bf16* yr = out + (size_t)r * CC;
  for (int c = threadIdx.x; c < CC; c += 128)
    yr[c] = (bf16)((xr[c] - mean) * rstd * gam[c] + bet[c]);
}

// ---------------------------------------------------------------------------
// WMMA GEMM: out[M,N] = A[M,K](bf16) @ W[N,K]^T(bf16) (+bias) (+epilogue)
// Block: 4 waves -> 64(M) x 64(N); wave: 32x32 (2x2 fragments).
// A and B k-slabs (64x32 each) double-buffered in LDS, filled by async DMA.
// ---------------------------------------------------------------------------
enum { EPI_BF16 = 0, EPI_BF16_GELU = 1, EPI_F32_SWIN = 2, EPI_F32_ADD = 3 };

template <int EPI>
__global__ __launch_bounds__(128)
void gemm_nt(const bf16* __restrict__ A, const bf16* __restrict__ W,
             const float* __restrict__ bias, void* __restrict__ out,
             const float* __restrict__ resid, int K, int N) {
  __shared__ __align__(16) bf16 Ab[2][64 * 32];
  __shared__ __align__(16) bf16 Bb[2][64 * 32];
  const int tid  = threadIdx.x;
  const int lane = tid & 31, wave = tid >> 5;
  const int l15  = lane & 15, half = lane >> 4;
  const int mblk = wave >> 1, nblk = wave & 1;
  const size_t Mbase = (size_t)blockIdx.x * 64;
  const int    Nbase = blockIdx.y * 64;

  const bf16* Ag = A + Mbase * K;
  const bf16* Wg = W + (size_t)Nbase * K;

  // stage one 64x32 k-slab of A and W into LDS buffer `buf`
  auto stage = [&](int buf, int kb) {
#pragma unroll
    for (int u = 0; u < 2; ++u) {
      int c = tid * 2 + u;                 // 256 16B-chunks per matrix
      int r = c >> 2, cc = (c & 3) * 8;
      async_cp16(Ag + (size_t)r * K + kb * 32 + cc, &Ab[buf][r * 32 + cc]);
      async_cp16(Wg + (size_t)r * K + kb * 32 + cc, &Bb[buf][r * 32 + cc]);
    }
  };

  v8f zz = {};
  v8f acc[2][2];
  acc[0][0] = zz; acc[0][1] = zz; acc[1][0] = zz; acc[1][1] = zz;

  const int nk = K / 32;
  stage(0, 0);
  for (int kb = 0; kb < nk; ++kb) {
    const int cur = kb & 1;
    async_wait_all();       // this wave's DMAs for tile kb have landed
    __syncthreads();        // -> all waves' DMAs have landed; prior reads done
    if (kb + 1 < nk) stage(cur ^ 1, kb + 1);

    V16 af[2], bw[2];
#pragma unroll
    for (int f = 0; f < 2; ++f) {
      const bf16* ap = &Ab[cur][(mblk * 32 + f * 16 + l15) * 32];
      af[f].h[0] = *reinterpret_cast<const v8bf*>(ap + half * 16);
      af[f].h[1] = *reinterpret_cast<const v8bf*>(ap + half * 16 + 8);
      const bf16* bp = &Bb[cur][(nblk * 32 + f * 16 + l15) * 32];
      bw[f].h[0] = *reinterpret_cast<const v8bf*>(bp + half * 16);
      bw[f].h[1] = *reinterpret_cast<const v8bf*>(bp + half * 16 + 8);
    }
#pragma unroll
    for (int fm = 0; fm < 2; ++fm)
#pragma unroll
      for (int fn = 0; fn < 2; ++fn)
        acc[fm][fn] = __builtin_amdgcn_wmma_f32_16x16x32_bf16(
            false, af[fm].v, false, bw[fn].v, (short)0, acc[fm][fn], false, false);
  }

#pragma unroll
  for (int fn = 0; fn < 2; ++fn) {
    const int col = Nbase + nblk * 32 + fn * 16 + l15;
    const float bv = bias ? bias[col] : 0.f;
#pragma unroll
    for (int fm = 0; fm < 2; ++fm) {
#pragma unroll
      for (int g = 0; g < 8; ++g) {
        int row = (int)Mbase + mblk * 32 + fm * 16 + g + half * 8;
        float v = acc[fm][fn][g] + bv;
        if (EPI == EPI_BF16) {
          ((bf16*)out)[(size_t)row * N + col] = (bf16)v;
        } else if (EPI == EPI_BF16_GELU) {
          v = 0.5f * v * (1.f + erff(v * 0.70710678118f));
          ((bf16*)out)[(size_t)row * N + col] = (bf16)v;
        } else if (EPI == EPI_F32_SWIN) {
          int t = swin_remap(row);
          ((float*)out)[(size_t)t * CC + col] = resid[(size_t)t * CC + col] + v;
        } else {  // EPI_F32_ADD
          ((float*)out)[(size_t)row * N + col] += v;
        }
      }
    }
  }
}

// ---------------------------------------------------------------------------
// Windowed attention: one wave per (window, head). N padded 49->64.
// ---------------------------------------------------------------------------
__global__ __launch_bounds__(128)
void attention_kernel(const bf16* __restrict__ qkv,
                      const float* __restrict__ bias,
                      bf16* __restrict__ obuf) {
  __shared__ __align__(16) bf16 Pl[4][64 * 64];   // P, row-major [n][m]
  __shared__ __align__(16) bf16 Vt[4][HD * 64];   // V^T, [d][key]
  const int wave = threadIdx.x >> 5, lane = threadIdx.x & 31;
  const int l15 = lane & 15, half = lane >> 4;
  const int pair = blockIdx.x * 4 + wave;
  const int win = pair / NHEADS, h = pair % NHEADS;
  const int nwin = win & 63;
  const bf16* base = qkv + (size_t)win * NWS * (3 * CC);
  const float* brow = bias + (size_t)(nwin * NHEADS + h) * 64 * 64;
  bf16* P = Pl[wave];
  bf16* V = Vt[wave];
  const float scale = 0.17677669529f;              // (C/NH)^-0.5

  for (int e = lane; e < 64 * HD; e += 32) {
    int key = e >> 5, d = e & 31;
    bf16 v = (bf16)0.f;
    if (key < NWS) v = base[(size_t)key * (3 * CC) + 2 * CC + h * HD + d];
    V[d * 64 + key] = v;
  }

  V16 kf[4];
#pragma unroll
  for (int j = 0; j < 4; ++j) {
    int key = j * 16 + l15;
    if (key < NWS) {
      const bf16* kr = base + (size_t)key * (3 * CC) + CC + h * HD + half * 16;
      kf[j].h[0] = load8g(kr);
      kf[j].h[1] = load8g(kr + 8);
    } else {
      v8bf z = {};
      kf[j].h[0] = z; kf[j].h[1] = z;
    }
  }

#pragma unroll
  for (int i = 0; i < 4; ++i) {
    V16 qf;
    {
      int qrow = i * 16 + l15;
      if (qrow < NWS) {
        const bf16* qr = base + (size_t)qrow * (3 * CC) + h * HD;
        qf.h[0] = load8g(qr + half * 16);
        qf.h[1] = load8g(qr + half * 16 + 8);
      } else {
        v8bf z = {};
        qf.h[0] = z; qf.h[1] = z;
      }
    }
    v8f s[4];
#pragma unroll
    for (int j = 0; j < 4; ++j) {
      v8f z = {};
      s[j] = __builtin_amdgcn_wmma_f32_16x16x32_bf16(
          false, qf.v, false, kf[j].v, (short)0, z, false, false);
    }
#pragma unroll
    for (int j = 0; j < 4; ++j)
#pragma unroll
      for (int g = 0; g < 8; ++g) {
        int nr = i * 16 + g + half * 8;
        s[j][g] = s[j][g] * scale + brow[nr * 64 + j * 16 + l15];
      }
#pragma unroll
    for (int g = 0; g < 8; ++g) {
      float mx = fmaxf(fmaxf(s[0][g], s[1][g]), fmaxf(s[2][g], s[3][g]));
      for (int d = 1; d < 16; d <<= 1) mx = fmaxf(mx, __shfl_xor(mx, d, 32));
      float e0 = __expf(s[0][g] - mx), e1 = __expf(s[1][g] - mx);
      float e2 = __expf(s[2][g] - mx), e3 = __expf(s[3][g] - mx);
      float sm = e0 + e1 + e2 + e3;
      for (int d = 1; d < 16; d <<= 1) sm += __shfl_xor(sm, d, 32);
      float r = 1.f / sm;
      int nr = (i * 16 + g + half * 8) * 64 + l15;
      P[nr +  0] = (bf16)(e0 * r);
      P[nr + 16] = (bf16)(e1 * r);
      P[nr + 32] = (bf16)(e2 * r);
      P[nr + 48] = (bf16)(e3 * r);
    }
  }
  __syncthreads();

#pragma unroll
  for (int i = 0; i < 4; ++i) {
    V16 pf[2];
#pragma unroll
    for (int kb = 0; kb < 2; ++kb) {
      const bf16* pr = P + (i * 16 + l15) * 64 + kb * 32 + half * 16;
      pf[kb].h[0] = *reinterpret_cast<const v8bf*>(pr);
      pf[kb].h[1] = *reinterpret_cast<const v8bf*>(pr + 8);
    }
#pragma unroll
    for (int nt = 0; nt < 2; ++nt) {
      v8f acc = {};
#pragma unroll
      for (int kb = 0; kb < 2; ++kb) {
        V16 vf;
        const bf16* vr = V + (nt * 16 + l15) * 64 + kb * 32 + half * 16;
        vf.h[0] = *reinterpret_cast<const v8bf*>(vr);
        vf.h[1] = *reinterpret_cast<const v8bf*>(vr + 8);
        acc = __builtin_amdgcn_wmma_f32_16x16x32_bf16(
            false, pf[kb].v, false, vf.v, (short)0, acc, false, false);
      }
#pragma unroll
      for (int g = 0; g < 8; ++g) {
        int nr = i * 16 + g + half * 8;
        if (nr < NWS)
          obuf[((size_t)win * NWS + nr) * CC + h * HD + nt * 16 + l15] =
              (bf16)acc[g];
      }
    }
  }
}

// ---------------------------------------------------------------------------
// host launcher
// ---------------------------------------------------------------------------
extern "C" void kernel_launch(void* const* d_in, const int* in_sizes, int n_in,
                              void* d_out, int out_size, void* d_ws, size_t ws_size,
                              hipStream_t stream) {
  (void)in_sizes; (void)n_in; (void)out_size; (void)ws_size;
  const float* x       = (const float*)d_in[0];
  const float* norm1_g = (const float*)d_in[1];
  const float* norm1_b = (const float*)d_in[2];
  const float* rpb     = (const float*)d_in[3];
  const float* qkv_v   = (const float*)d_in[4];
  const float* qkv_u   = (const float*)d_in[5];
  const float* qkv_b   = (const float*)d_in[6];
  const float* proj_v  = (const float*)d_in[7];
  const float* proj_u  = (const float*)d_in[8];
  const float* proj_b  = (const float*)d_in[9];
  const float* norm2_g = (const float*)d_in[10];
  const float* norm2_b = (const float*)d_in[11];
  const float* fc1_v   = (const float*)d_in[12];
  const float* fc1_u   = (const float*)d_in[13];
  const float* fc1_b   = (const float*)d_in[14];
  const float* fc2_v   = (const float*)d_in[15];
  const float* fc2_u   = (const float*)d_in[16];
  const float* fc2_b   = (const float*)d_in[17];
  float* out = (float*)d_out;

  char* p = (char*)d_ws;
  auto alloc = [&](size_t bytes) -> void* {
    void* r = (void*)p;
    p += (bytes + 255) & ~(size_t)255;
    return r;
  };
  bf16* wqv  = (bf16*)alloc((size_t)RANK * CC * 2);
  bf16* wqu  = (bf16*)alloc((size_t)3 * CC * RANK * 2);
  bf16* wpv  = (bf16*)alloc((size_t)RANK * CC * 2);
  bf16* wpu  = (bf16*)alloc((size_t)CC * RANK * 2);
  bf16* wf1v = (bf16*)alloc((size_t)RANK * CC * 2);
  bf16* wf1u = (bf16*)alloc((size_t)HID * RANK * 2);
  bf16* wf2v = (bf16*)alloc((size_t)RANK * HID * 2);
  bf16* wf2u = (bf16*)alloc((size_t)CC * RANK * 2);
  float* abias = (float*)alloc((size_t)64 * NHEADS * 64 * 64 * 4);
  bf16* yw   = (bf16*)alloc((size_t)MROWS * CC * 2);
  bf16* tbuf = (bf16*)alloc((size_t)MROWS * RANK * 2);
  bf16* big  = (bf16*)alloc((size_t)MROWS * HID * 2);
  bf16* obuf = (bf16*)alloc((size_t)MROWS * CC * 2);

  auto cvt = [&](const float* s, bf16* d, int n) {
    cvt_bf16<<<(n + 255) / 256, 256, 0, stream>>>(s, d, n);
  };
  cvt(qkv_v,  wqv,  RANK * CC);
  cvt(qkv_u,  wqu,  3 * CC * RANK);
  cvt(proj_v, wpv,  RANK * CC);
  cvt(proj_u, wpu,  CC * RANK);
  cvt(fc1_v,  wf1v, RANK * CC);
  cvt(fc1_u,  wf1u, HID * RANK);
  cvt(fc2_v,  wf2v, RANK * HID);
  cvt(fc2_u,  wf2u, CC * RANK);

  attn_bias_kernel<<<64 * NHEADS, 256, 0, stream>>>(rpb, abias);
  ln_kernel<true><<<MROWS, 128, 0, stream>>>(x, norm1_g, norm1_b, yw);

  const int MB = MROWS / 64;  // 1568
  gemm_nt<EPI_BF16><<<dim3(MB, RANK / 64), 128, 0, stream>>>(
      yw, wqv, nullptr, tbuf, nullptr, CC, RANK);
  gemm_nt<EPI_BF16><<<dim3(MB, (3 * CC) / 64), 128, 0, stream>>>(
      tbuf, wqu, qkv_b, big, nullptr, RANK, 3 * CC);

  attention_kernel<<<(2048 * NHEADS) / 4, 128, 0, stream>>>(big, abias, obuf);

  gemm_nt<EPI_BF16><<<dim3(MB, RANK / 64), 128, 0, stream>>>(
      obuf, wpv, nullptr, tbuf, nullptr, CC, RANK);
  gemm_nt<EPI_F32_SWIN><<<dim3(MB, CC / 64), 128, 0, stream>>>(
      tbuf, wpu, proj_b, out, x, RANK, CC);

  ln_kernel<false><<<MROWS, 128, 0, stream>>>(out, norm2_g, norm2_b, yw);

  gemm_nt<EPI_BF16><<<dim3(MB, RANK / 64), 128, 0, stream>>>(
      yw, wf1v, nullptr, tbuf, nullptr, CC, RANK);
  gemm_nt<EPI_BF16_GELU><<<dim3(MB, HID / 64), 128, 0, stream>>>(
      tbuf, wf1u, fc1_b, big, nullptr, RANK, HID);
  gemm_nt<EPI_BF16><<<dim3(MB, RANK / 64), 128, 0, stream>>>(
      big, wf2v, nullptr, tbuf, nullptr, HID, RANK);
  gemm_nt<EPI_F32_ADD><<<dim3(MB, CC / 64), 128, 0, stream>>>(
      tbuf, wf2u, fc2_b, out, nullptr, RANK, CC);
}
